// SmoothTripletHead_30185030156998
// MI455X (gfx1250) — matrix operand order repaired and checked by reading, not compiled, not run on hardware
//
#include <hip/hip_runtime.h>
#include <hip/hip_bf16.h>
#include <math.h>

// ---------------------------------------------------------------------------
// SmoothTripletHead fused pipeline for MI455X (gfx1250, wave32, WMMA)
//   N = 8192 rows, C = 256 features, K = 5 hardest negatives,
//   gamma = 1, margin = 2
// ---------------------------------------------------------------------------

#define N_ROWS   8192
#define N_FEAT   256
#define TOPK     5
#define MARGIN   2.0f
#define N_SLICES 4                 // column slices (grid.y of gemm kernel)
#define SLICE_COLS (N_ROWS / N_SLICES)     // 2048
#define SLICE_TILES (SLICE_COLS / 16)      // 128
#define ROWS_PER_BLOCK 128                 // 8 waves * 16 rows

typedef __attribute__((ext_vector_type(16))) _Float16 v16h;
typedef __attribute__((ext_vector_type(8)))  _Float16 v8h;
typedef __attribute__((ext_vector_type(8)))  float    v8f;
typedef __attribute__((ext_vector_type(4)))  float    v4f;

static __device__ inline v16h cat8(v8h a, v8h b) {
    v16h r;
#pragma unroll
    for (int i = 0; i < 8; ++i) { r[i] = a[i]; r[8 + i] = b[i]; }
    return r;
}

// Branchless sorted-ascending top-5 insert:
//   r_j = min(s_{j+1}, max(s_j, v)),  s_5 = +inf  -> 9 min/max ops, no branches
static __device__ inline void insert5_nb(float (&s)[5], float v) {
    float r0 = fminf(s[1], fmaxf(s[0], v));
    float r1 = fminf(s[2], fmaxf(s[1], v));
    float r2 = fminf(s[3], fmaxf(s[2], v));
    float r3 = fminf(s[4], fmaxf(s[3], v));
    float r4 = fmaxf(s[4], v);
    s[0] = r0; s[1] = r1; s[2] = r2; s[3] = r3; s[4] = r4;
}

static __device__ inline v4f vmax4(v4f a, v4f b) {
    v4f r;
#pragma unroll
    for (int i = 0; i < 4; ++i) r[i] = fmaxf(a[i], b[i]);
    return r;
}

// ---------------------------------------------------------------------------
// Kernel 1: row-normalize fp32 -> f16   (grid: (N_ROWS, 2), block: 256)
// ---------------------------------------------------------------------------
__global__ __launch_bounds__(256) void k_norm(const float* __restrict__ in,
                                              const float* __restrict__ tgt,
                                              _Float16* __restrict__ Af,
                                              _Float16* __restrict__ Bf) {
    const float*  src = (blockIdx.y == 0) ? in : tgt;
    _Float16*     dst = (blockIdx.y == 0) ? Af : Bf;
    const int row = blockIdx.x;
    const int t   = threadIdx.x;

    float x  = src[(size_t)row * N_FEAT + t];
    float ss = x * x;
#pragma unroll
    for (int o = 16; o > 0; o >>= 1) ss += __shfl_xor(ss, o, 32);

    __shared__ float part[8];
    if ((t & 31) == 0) part[t >> 5] = ss;
    __syncthreads();
    float tot = 0.f;
#pragma unroll
    for (int i = 0; i < 8; ++i) tot += part[i];

    float rn = rsqrtf(tot);
    dst[(size_t)row * N_FEAT + t] = (_Float16)(x * rn);
}

// ---------------------------------------------------------------------------
// Kernel 2: fp32 diagonal cosine sim    (grid: N_ROWS, block: 256)
// ---------------------------------------------------------------------------
__global__ __launch_bounds__(256) void k_diag(const float* __restrict__ in,
                                              const float* __restrict__ tgt,
                                              float* __restrict__ diag) {
    const int row = blockIdx.x;
    const int t   = threadIdx.x;
    float a = in [(size_t)row * N_FEAT + t];
    float b = tgt[(size_t)row * N_FEAT + t];
    float d = a * b, na = a * a, nb = b * b;
#pragma unroll
    for (int o = 16; o > 0; o >>= 1) {
        d  += __shfl_xor(d,  o, 32);
        na += __shfl_xor(na, o, 32);
        nb += __shfl_xor(nb, o, 32);
    }
    __shared__ float pd[8], pa[8], pb[8];
    if ((t & 31) == 0) { pd[t >> 5] = d; pa[t >> 5] = na; pb[t >> 5] = nb; }
    __syncthreads();
    if (t == 0) {
        float D = 0.f, A = 0.f, B = 0.f;
#pragma unroll
        for (int i = 0; i < 8; ++i) { D += pd[i]; A += pa[i]; B += pb[i]; }
        diag[row] = D * rsqrtf(A * B);
    }
}

// ---------------------------------------------------------------------------
// Kernel 3: fused WMMA similarity GEMM + per-row top-5 (per column slice)
//   grid: (N_ROWS/128, N_SLICES), block: 256 (8 wave32 waves)
//   top5 layout: top5[(slice*N_ROWS + row)*TOPK + j], ascending
// ---------------------------------------------------------------------------
__global__ __launch_bounds__(256) void k_gemm_topk(const _Float16* __restrict__ A,
                                                   const _Float16* __restrict__ B,
                                                   float* __restrict__ top5) {
    __shared__ _Float16 ldsB[16 * 264];       // B tile, padded stride (bank-safe)
    __shared__ float    ldsS[8][16 * 20];     // per-wave accum transpose buffer

    const int tid  = threadIdx.x;
    const int wave = tid >> 5;
    const int lane = tid & 31;
    const int m    = lane & 15;               // row within wave strip / col idx
    const int h    = lane >> 4;               // lane half
    const int rowbase = blockIdx.x * ROWS_PER_BLOCK + wave * 16;
    const int gr      = rowbase + m;          // this lane's matrix row

    // ---- preload the wave's A fragments (16 rows x 256 K) into registers ----
    // ISA 16-bit A 16x32 layout: lane(h,m) holds K = ks*32 + h*8 + [0..7]
    //                            and           K = ks*32 + 16 + h*8 + [0..7]
    v16h af[8];
    {
        const _Float16* arow = A + (size_t)gr * N_FEAT;
#pragma unroll
        for (int ks = 0; ks < 8; ++ks) {
            v8h c0 = *(const v8h*)(arow + ks * 32 + h * 8);
            v8h c1 = *(const v8h*)(arow + ks * 32 + 16 + h * 8);
            af[ks] = cat8(c0, c1);
        }
    }

    float s[TOPK];
#pragma unroll
    for (int j = 0; j < TOPK; ++j) s[j] = -INFINITY;

    const int c0base = blockIdx.y * SLICE_COLS;
    float* const sm = &ldsS[wave][0];
    const float* const rowp = sm + ((lane & 15) * 20);
    const _Float16* const bbase = ldsB + m * 264 + h * 8;

    // B 32x16 f16 layout: lane(h,n=m) holds K = ks*32 + h*8 + [0..7] and ks*32 + 16 + h*8 + [0..7]
    auto loadB = [&](int ks) -> v16h {
        const _Float16* bp = bbase + ks * 32;
        v8h b0 = *(const v8h*)bp;
        v8h b1 = *(const v8h*)(bp + 16);
        return cat8(b0, b1);
    };

    for (int ct = 0; ct < SLICE_TILES; ++ct) {
        const int c0 = c0base + ct * 16;

        __syncthreads();
        // ---- cooperative stage of B tile: 16 target rows x 256 K (8 KB) ----
        {
            const int lr = tid >> 4;          // 0..15 tile row
            const int ch = tid & 15;          // 0..15 chunk of 16 f16 (32 B)
            const uint4* src = (const uint4*)(B + (size_t)(c0 + lr) * N_FEAT + ch * 16);
            uint4*       dst = (uint4*)(ldsB + lr * 264 + ch * 16);
            dst[0] = src[0];
            dst[1] = src[1];
            if (ct + 1 < SLICE_TILES)         // L2 prefetch of next tile
                __builtin_prefetch(B + (size_t)(c0 + 16 + lr) * N_FEAT + ch * 16, 0, 3);
        }
        __syncthreads();

        // ---- software-pipelined WMMA over K = 256 ----
        // two rotating B buffers (loads run ahead of consuming WMMA) and two
        // accumulators (no single serial RAW chain on C)
        v8f acc0 = {0.f, 0.f, 0.f, 0.f, 0.f, 0.f, 0.f, 0.f};
        v8f acc1 = {0.f, 0.f, 0.f, 0.f, 0.f, 0.f, 0.f, 0.f};
        v16h b0 = loadB(0);
        v16h b1 = loadB(1);
        acc0 = __builtin_amdgcn_wmma_f32_16x16x32_f16(false, af[0], false, b0, (short)0, acc0, false, false);
        b0 = loadB(2);
        acc1 = __builtin_amdgcn_wmma_f32_16x16x32_f16(false, af[1], false, b1, (short)0, acc1, false, false);
        b1 = loadB(3);
        acc0 = __builtin_amdgcn_wmma_f32_16x16x32_f16(false, af[2], false, b0, (short)0, acc0, false, false);
        b0 = loadB(4);
        acc1 = __builtin_amdgcn_wmma_f32_16x16x32_f16(false, af[3], false, b1, (short)0, acc1, false, false);
        b1 = loadB(5);
        acc0 = __builtin_amdgcn_wmma_f32_16x16x32_f16(false, af[4], false, b0, (short)0, acc0, false, false);
        b0 = loadB(6);
        acc1 = __builtin_amdgcn_wmma_f32_16x16x32_f16(false, af[5], false, b1, (short)0, acc1, false, false);
        b1 = loadB(7);
        acc0 = __builtin_amdgcn_wmma_f32_16x16x32_f16(false, af[6], false, b0, (short)0, acc0, false, false);
        acc1 = __builtin_amdgcn_wmma_f32_16x16x32_f16(false, af[7], false, b1, (short)0, acc1, false, false);

        v8f acc;
#pragma unroll
        for (int i = 0; i < 8; ++i) acc[i] = acc0[i] + acc1[i];

        // ---- transpose accumulator through per-wave LDS ----
        // C/D layout: VGPR r -> M = r + h*8, N = lane&15
#pragma unroll
        for (int r = 0; r < 8; ++r) sm[(r + h * 8) * 20 + m] = acc[r];
        __builtin_amdgcn_wave_barrier();      // LDS is in-order within a wave

        // ---- lanes 0..15 each own one row: update register top-5 ----
        if (lane < 16) {
            // mask this row's diagonal element in place (same-lane LDS: in-order)
            const int dn = gr - c0;
            if ((unsigned)dn < 16u) sm[lane * 20 + dn] = -INFINITY;

            v4f q0 = *(const v4f*)(rowp + 0);
            v4f q1 = *(const v4f*)(rowp + 4);
            v4f q2 = *(const v4f*)(rowp + 8);
            v4f q3 = *(const v4f*)(rowp + 12);

            v4f qm = vmax4(vmax4(q0, q1), vmax4(q2, q3));
            float mx = fmaxf(fmaxf(qm[0], qm[1]), fmaxf(qm[2], qm[3]));

            if (mx > s[0]) {                  // tile contributes a new top-5 entry
#pragma unroll
                for (int i = 0; i < 4; ++i) insert5_nb(s, q0[i]);
#pragma unroll
                for (int i = 0; i < 4; ++i) insert5_nb(s, q1[i]);
#pragma unroll
                for (int i = 0; i < 4; ++i) insert5_nb(s, q2[i]);
#pragma unroll
                for (int i = 0; i < 4; ++i) insert5_nb(s, q3[i]);
            }
        }
        __builtin_amdgcn_wave_barrier();
    }

    if (lane < 16) {
        float* o = top5 + ((size_t)blockIdx.y * N_ROWS + gr) * TOPK;
#pragma unroll
        for (int j = 0; j < TOPK; ++j) o[j] = s[j];
    }
}

// ---------------------------------------------------------------------------
// Kernel 4: merge slice top-5 lists, compute per-row margin loss sum
//   grid: N_ROWS/256, block 256
// ---------------------------------------------------------------------------
__global__ __launch_bounds__(256) void k_rowloss(const float* __restrict__ top5,
                                                 const float* __restrict__ diag,
                                                 float* __restrict__ rowloss) {
    const int r = blockIdx.x * 256 + threadIdx.x;
    float s[TOPK];
#pragma unroll
    for (int j = 0; j < TOPK; ++j) s[j] = -INFINITY;
#pragma unroll
    for (int sl = 0; sl < N_SLICES; ++sl) {
        const float* p = top5 + ((size_t)sl * N_ROWS + r) * TOPK;
#pragma unroll
        for (int j = 0; j < TOPK; ++j) insert5_nb(s, p[j]);
    }
    const float d = diag[r];
    float acc = 0.f;
    // loss_j = relu(gamma*(-d) - (-s_j) + margin) = relu(s_j - d + 2)
#pragma unroll
    for (int j = 0; j < TOPK; ++j) acc += fmaxf(0.f, s[j] - d + MARGIN);
    rowloss[r] = acc;
}

// ---------------------------------------------------------------------------
// Kernel 5: deterministic final reduction -> scalar mean
// ---------------------------------------------------------------------------
__global__ __launch_bounds__(256) void k_reduce(const float* __restrict__ rowloss,
                                                float* __restrict__ out) {
    float sum = 0.f;
    for (int i = threadIdx.x; i < N_ROWS; i += 256) sum += rowloss[i];
#pragma unroll
    for (int o = 16; o > 0; o >>= 1) sum += __shfl_xor(sum, o, 32);
    __shared__ float part[8];
    if ((threadIdx.x & 31) == 0) part[threadIdx.x >> 5] = sum;
    __syncthreads();
    if (threadIdx.x == 0) {
        float t = 0.f;
#pragma unroll
        for (int i = 0; i < 8; ++i) t += part[i];
        out[0] = t / ((float)N_ROWS * (float)TOPK);
    }
}

// ---------------------------------------------------------------------------
extern "C" void kernel_launch(void* const* d_in, const int* in_sizes, int n_in,
                              void* d_out, int out_size, void* d_ws, size_t ws_size,
                              hipStream_t stream) {
    const float* in  = (const float*)d_in[0];
    const float* tgt = (const float*)d_in[1];
    float* out = (float*)d_out;

    // workspace carve-up (~8.7 MB)
    char* w = (char*)d_ws;
    _Float16* Af   = (_Float16*)(w);                                   // 4 MB
    _Float16* Bf   = (_Float16*)(w + ((size_t)4 << 20));               // 4 MB
    float*    diag = (float*)(w + ((size_t)8 << 20));                  // 32 KB
    float*    top5 = (float*)(w + ((size_t)8 << 20) + ((size_t)32 << 10));   // 640 KB
    float*    rowl = (float*)(w + ((size_t)8 << 20) + ((size_t)704 << 10));  // 32 KB

    k_norm<<<dim3(N_ROWS, 2), 256, 0, stream>>>(in, tgt, Af, Bf);
    k_diag<<<dim3(N_ROWS), 256, 0, stream>>>(in, tgt, diag);
    k_gemm_topk<<<dim3(N_ROWS / ROWS_PER_BLOCK, N_SLICES), 256, 0, stream>>>(Af, Bf, top5);
    k_rowloss<<<dim3(N_ROWS / 256), 256, 0, stream>>>(top5, diag, rowl);
    k_reduce<<<dim3(1), 256, 0, stream>>>(rowl, out);
}
